// PositionProbeTransformer_28157805593366
// MI455X (gfx1250) — compile-verified, matrix operand order checked by reading, not run on hardware
//
#include <hip/hip_runtime.h>

// ---------------------------------------------------------------------------
// PositionProbeTransformer forward for MI455X (gfx1250), wave32 + WMMA bf16.
// GEMMs: v_wmma_f32_16x16x32_bf16, 32x64 per-wave tiles, K-pipelined loads.
// Embedding gather: async global->LDS->global DMA path (ASYNCcnt).
// ---------------------------------------------------------------------------

typedef unsigned short u16;
typedef __attribute__((ext_vector_type(16))) __bf16       v16bf;
typedef __attribute__((ext_vector_type(8)))  float        v8f;
typedef __attribute__((ext_vector_type(4)))  unsigned int u32x4;

union FragBf { v16bf v; u32x4 q[2]; };

enum { BATCH = 4, SEQ = 1024, DIM = 1024, HEADS = 16, DH = 64, HALF = 32,
       FF = 4096, LAYERS = 6, MROWS = BATCH * SEQ };

__device__ __forceinline__ u16 f2bf(float f) {
  unsigned int u = __builtin_bit_cast(unsigned int, f);
  u += 0x7FFFu + ((u >> 16) & 1u);           // round to nearest even
  return (u16)(u >> 16);
}

__device__ __forceinline__ v8f vzero() {
  v8f z;
#pragma unroll
  for (int r = 0; r < 8; ++r) z[r] = 0.0f;
  return z;
}

// Per the CDNA5 16-bit A/B layout, lanes 0-15 carry K 0..7 and 16..23; lanes
// 16-31 carry K 8..15 and 24..31 -> two contiguous 16-byte loads per lane.
__device__ __forceinline__ FragBf load_frag(const u16* base, int hi) {
  FragBf f;
  f.q[0] = *(const u32x4*)(base + hi * 8);
  f.q[1] = *(const u32x4*)(base + hi * 8 + 16);
  return f;
}

__device__ __forceinline__ v8f wmma_bf16(const FragBf& a, const FragBf& b, v8f c) {
  return __builtin_amdgcn_wmma_f32_16x16x32_bf16(false, a.v, false, b.v,
                                                 (short)0, c, false, false);
}

// ---------------------------------------------------------------------------
// Embedding gather via async LDS DMA: tok_emb row -> LDS -> h row. Each of the
// 256 lanes moves 16B; data never passes through VGPRs (ASYNCcnt tracked).
// ---------------------------------------------------------------------------
__global__ __launch_bounds__(256) void k_embed(const int* __restrict__ tokens,
                                               const float* __restrict__ emb,
                                               float* __restrict__ h) {
  __shared__ float tile[DIM];                      // 4 KB staging
  int row = blockIdx.x;
  int tok = tokens[row];
  int t = threadIdx.x;                             // 0..255, 16B per lane

  unsigned lds = (unsigned)(size_t)(void*)(tile + t * 4);
  unsigned long long gsrc = (unsigned long long)(size_t)(emb + (size_t)tok * DIM + t * 4);
  unsigned long long gdst = (unsigned long long)(size_t)(h + (size_t)row * DIM + t * 4);

  asm volatile("global_load_async_to_lds_b128 %0, %1, off"
               :: "v"(lds), "v"(gsrc) : "memory");
  asm volatile("s_wait_asynccnt 0x0" ::: "memory");
  asm volatile("global_store_async_from_lds_b128 %0, %1, off"
               :: "v"(gdst), "v"(lds) : "memory");
  asm volatile("s_wait_asynccnt 0x0" ::: "memory");
}

// ---------------------------------------------------------------------------
// LayerNorm over D=1024; optional bf16 and f32 outputs.
// ---------------------------------------------------------------------------
__global__ __launch_bounds__(256) void k_layernorm(const float* __restrict__ x,
                                                   const float* __restrict__ sc,
                                                   const float* __restrict__ bi,
                                                   u16* __restrict__ out_bf,
                                                   float* __restrict__ out_f32) {
  int row = blockIdx.x;
  const float* xr = x + (size_t)row * DIM;
  __shared__ float red[256];
  float sum = 0.f, sq = 0.f;
  for (int i = threadIdx.x; i < DIM; i += 256) {
    float v = xr[i]; sum += v; sq += v * v;
  }
  red[threadIdx.x] = sum; __syncthreads();
  for (int o = 128; o > 0; o >>= 1) {
    if (threadIdx.x < o) red[threadIdx.x] += red[threadIdx.x + o];
    __syncthreads();
  }
  float mean = red[0] * (1.0f / DIM);
  __syncthreads();
  red[threadIdx.x] = sq; __syncthreads();
  for (int o = 128; o > 0; o >>= 1) {
    if (threadIdx.x < o) red[threadIdx.x] += red[threadIdx.x + o];
    __syncthreads();
  }
  float var = red[0] * (1.0f / DIM) - mean * mean;
  float inv = rsqrtf(var + 1e-5f);
  for (int i = threadIdx.x; i < DIM; i += 256) {
    float v = (xr[i] - mean) * inv * sc[i] + bi[i];
    if (out_bf)  out_bf[(size_t)row * DIM + i]  = f2bf(v);
    if (out_f32) out_f32[(size_t)row * DIM + i] = v;
  }
}

// ---------------------------------------------------------------------------
// Weight convert + transpose: W (K x N) f32 -> Wt (N x K) bf16
// ---------------------------------------------------------------------------
__global__ __launch_bounds__(256) void k_convert(const float* __restrict__ W,
                                                 u16* __restrict__ Wt,
                                                 int K, int N) {
  int idx = blockIdx.x * 256 + threadIdx.x;
  if (idx >= K * N) return;
  int k = idx / N, n = idx - k * N;
  Wt[(size_t)n * K + k] = f2bf(W[idx]);
}

// ---------------------------------------------------------------------------
// bf16 GEMM: C(MxN) = A(MxK) @ Bt(NxK)^T [+ Cadd]. One wave per block computes
// a 32x64 strip: 2 A-fragments x 4 B-fragments -> 8 WMMA accumulators, with
// the K-loop software-pipelined (next step's 12 b128 loads issued before the
// current step's 8 WMMAs) plus prefetch 256B ahead on each stream.
// grid = (N/64, M/32), block = 32.
// ---------------------------------------------------------------------------
__global__ __launch_bounds__(32) void k_gemm(const u16* __restrict__ A,
                                             const u16* __restrict__ Bt,
                                             const float* __restrict__ Cadd,
                                             float* __restrict__ C,
                                             int M, int N, int K) {
  int nb = blockIdx.x * 64;
  int mb = blockIdx.y * 32;
  int lane = threadIdx.x;
  int row = lane & 15;
  int hi = lane >> 4;

  v8f acc[2][4];
#pragma unroll
  for (int u = 0; u < 2; ++u)
#pragma unroll
    for (int t = 0; t < 4; ++t) {
      acc[u][t] = vzero();
      if (Cadd) {
#pragma unroll
        for (int r = 0; r < 8; ++r)
          acc[u][t][r] =
              Cadd[(size_t)(mb + u * 16 + hi * 8 + r) * N + nb + t * 16 + row];
      }
    }

  const u16* arow0 = A + (size_t)(mb + row) * K;
  const u16* arow1 = A + (size_t)(mb + 16 + row) * K;
  const u16* brow0 = Bt + (size_t)(nb + row) * K;
  const u16* brow1 = Bt + (size_t)(nb + 16 + row) * K;
  const u16* brow2 = Bt + (size_t)(nb + 32 + row) * K;
  const u16* brow3 = Bt + (size_t)(nb + 48 + row) * K;

  FragBf a0 = load_frag(arow0, hi), a1 = load_frag(arow1, hi);
  FragBf b0 = load_frag(brow0, hi), b1 = load_frag(brow1, hi);
  FragBf b2 = load_frag(brow2, hi), b3 = load_frag(brow3, hi);

  for (int kk = 32; kk < K; kk += 32) {
    __builtin_prefetch(arow0 + kk + 128, 0, 1);
    __builtin_prefetch(brow0 + kk + 128, 0, 1);
    __builtin_prefetch(brow2 + kk + 128, 0, 1);
    FragBf na0 = load_frag(arow0 + kk, hi);
    FragBf na1 = load_frag(arow1 + kk, hi);
    FragBf nb0 = load_frag(brow0 + kk, hi);
    FragBf nb1 = load_frag(brow1 + kk, hi);
    FragBf nb2 = load_frag(brow2 + kk, hi);
    FragBf nb3 = load_frag(brow3 + kk, hi);

    acc[0][0] = wmma_bf16(a0, b0, acc[0][0]);
    acc[1][0] = wmma_bf16(a1, b0, acc[1][0]);
    acc[0][1] = wmma_bf16(a0, b1, acc[0][1]);
    acc[1][1] = wmma_bf16(a1, b1, acc[1][1]);
    acc[0][2] = wmma_bf16(a0, b2, acc[0][2]);
    acc[1][2] = wmma_bf16(a1, b2, acc[1][2]);
    acc[0][3] = wmma_bf16(a0, b3, acc[0][3]);
    acc[1][3] = wmma_bf16(a1, b3, acc[1][3]);

    a0 = na0; a1 = na1; b0 = nb0; b1 = nb1; b2 = nb2; b3 = nb3;
  }

  acc[0][0] = wmma_bf16(a0, b0, acc[0][0]);
  acc[1][0] = wmma_bf16(a1, b0, acc[1][0]);
  acc[0][1] = wmma_bf16(a0, b1, acc[0][1]);
  acc[1][1] = wmma_bf16(a1, b1, acc[1][1]);
  acc[0][2] = wmma_bf16(a0, b2, acc[0][2]);
  acc[1][2] = wmma_bf16(a1, b2, acc[1][2]);
  acc[0][3] = wmma_bf16(a0, b3, acc[0][3]);
  acc[1][3] = wmma_bf16(a1, b3, acc[1][3]);

#pragma unroll
  for (int u = 0; u < 2; ++u)
#pragma unroll
    for (int t = 0; t < 4; ++t)
#pragma unroll
      for (int r = 0; r < 8; ++r)
        C[(size_t)(mb + u * 16 + hi * 8 + r) * N + nb + t * 16 + row] =
            acc[u][t][r];
}

// ---------------------------------------------------------------------------
// RoPE + head split. Reads qkv f32 (M x 3D). Writes:
//   q_bf, k_bf : (B,H,S,DH) bf16 row-major   (rope applied, interleaved pairs)
//   v_t        : (B,H,DH,S) bf16             (transposed for probs@V WMMA)
// ---------------------------------------------------------------------------
__global__ __launch_bounds__(256) void k_rope_split(const float* __restrict__ qkv,
                                                    u16* __restrict__ qb,
                                                    u16* __restrict__ kb,
                                                    u16* __restrict__ vt) {
  int tid = blockIdx.x * 256 + threadIdx.x;   // B*S*H*HALF = 2^21 threads
  int i = tid & 31;
  int h = (tid >> 5) & 15;
  int s = (tid >> 9) & 1023;
  int b = tid >> 19;

  const float LOG_BASE = 9.210340371976184f;  // ln(10000)
  float inv_freq = __expf(-(2.0f * i) * (LOG_BASE / 64.0f));
  float ang = (float)s * inv_freq;
  float c = __cosf(ang), sn = __sinf(ang);

  size_t rowbase = ((size_t)(b * SEQ + s)) * (3 * DIM);
  int col = h * DH + 2 * i;

  float q0 = qkv[rowbase + col],           q1 = qkv[rowbase + col + 1];
  float k0 = qkv[rowbase + DIM + col],     k1 = qkv[rowbase + DIM + col + 1];
  float v0 = qkv[rowbase + 2 * DIM + col], v1 = qkv[rowbase + 2 * DIM + col + 1];

  size_t hdst = (((size_t)(b * HEADS + h)) * SEQ + s) * DH + 2 * i;
  qb[hdst]     = f2bf(q0 * c - q1 * sn);
  qb[hdst + 1] = f2bf(q1 * c + q0 * sn);
  kb[hdst]     = f2bf(k0 * c - k1 * sn);
  kb[hdst + 1] = f2bf(k1 * c + k0 * sn);

  size_t vbase = ((size_t)(b * HEADS + h)) * DH * SEQ;
  vt[vbase + (size_t)(2 * i) * SEQ + s]     = f2bf(v0);
  vt[vbase + (size_t)(2 * i + 1) * SEQ + s] = f2bf(v1);
}

// ---------------------------------------------------------------------------
// Attention logits: up = q[:, :32]@k[:, :32]^T, dn = q[:, 32:]@k[:, 32:]^T,
// logits = (qi >= kj ? dn : up) * 32^-0.5. grid = (S/64, S/16, B*H), block=32.
// ---------------------------------------------------------------------------
__global__ __launch_bounds__(32) void k_scores(const u16* __restrict__ q,
                                               const u16* __restrict__ k,
                                               float* __restrict__ scores) {
  int bh = blockIdx.z;
  int mb = blockIdx.y * 16;
  int nb = blockIdx.x * 64;
  int lane = threadIdx.x, row = lane & 15, hi = lane >> 4;

  const u16* qbase = q + (size_t)bh * SEQ * DH;
  const u16* kbase = k + (size_t)bh * SEQ * DH;

  FragBf aup = load_frag(qbase + (size_t)(mb + row) * DH + 0,  hi);
  FragBf adn = load_frag(qbase + (size_t)(mb + row) * DH + 32, hi);

  v8f up[4], dn[4];
#pragma unroll
  for (int t = 0; t < 4; ++t) { up[t] = vzero(); dn[t] = vzero(); }

#pragma unroll
  for (int t = 0; t < 4; ++t) {
    FragBf bu = load_frag(kbase + (size_t)(nb + t * 16 + row) * DH + 0,  hi);
    FragBf bd = load_frag(kbase + (size_t)(nb + t * 16 + row) * DH + 32, hi);
    up[t] = wmma_bf16(aup, bu, up[t]);
    dn[t] = wmma_bf16(adn, bd, dn[t]);
  }

  const float scale = 0.17677669529663687f;   // 1/sqrt(32)
#pragma unroll
  for (int t = 0; t < 4; ++t)
#pragma unroll
    for (int r = 0; r < 8; ++r) {
      int qi = mb + hi * 8 + r;
      int kj = nb + t * 16 + row;
      float v = (qi >= kj ? dn[t][r] : up[t][r]) * scale;
      scores[((size_t)bh * SEQ + qi) * SEQ + kj] = v;
    }
}

// ---------------------------------------------------------------------------
// Row softmax over S=1024 keys; emit bf16 probabilities. grid = B*H*S.
// ---------------------------------------------------------------------------
__global__ __launch_bounds__(256) void k_softmax_probs(const float* __restrict__ s,
                                                       u16* __restrict__ p) {
  size_t row = blockIdx.x;
  const float* sr = s + row * SEQ;
  __shared__ float red[256];
  float vals[4];
  float m = -1e30f;
#pragma unroll
  for (int j = 0; j < 4; ++j) {
    vals[j] = sr[threadIdx.x + j * 256];
    m = fmaxf(m, vals[j]);
  }
  red[threadIdx.x] = m; __syncthreads();
  for (int o = 128; o > 0; o >>= 1) {
    if (threadIdx.x < o) red[threadIdx.x] = fmaxf(red[threadIdx.x], red[threadIdx.x + o]);
    __syncthreads();
  }
  float rmax = red[0]; __syncthreads();
  float sum = 0.f;
#pragma unroll
  for (int j = 0; j < 4; ++j) { vals[j] = __expf(vals[j] - rmax); sum += vals[j]; }
  red[threadIdx.x] = sum; __syncthreads();
  for (int o = 128; o > 0; o >>= 1) {
    if (threadIdx.x < o) red[threadIdx.x] += red[threadIdx.x + o];
    __syncthreads();
  }
  float inv = 1.0f / red[0];
#pragma unroll
  for (int j = 0; j < 4; ++j)
    p[row * SEQ + threadIdx.x + j * 256] = f2bf(vals[j] * inv);
}

// ---------------------------------------------------------------------------
// o = probs @ V via WMMA, scattering head-merged bf16 output (B*S, D).
// 32x64 per-wave tile like k_gemm. grid = (S/32, B*H), block = 32.
// ---------------------------------------------------------------------------
__global__ __launch_bounds__(32) void k_attn_out(const u16* __restrict__ probs,
                                                 const u16* __restrict__ vt,
                                                 u16* __restrict__ obf) {
  int bh = blockIdx.y;
  int b = bh >> 4, hd = bh & 15;
  int mb = blockIdx.x * 32;
  int lane = threadIdx.x, row = lane & 15, hi = lane >> 4;

  const u16* pr = probs + (size_t)bh * SEQ * SEQ;
  const u16* vb = vt + (size_t)bh * DH * SEQ;

  v8f acc[2][4];
#pragma unroll
  for (int u = 0; u < 2; ++u)
#pragma unroll
    for (int t = 0; t < 4; ++t) acc[u][t] = vzero();

  const u16* arow0 = pr + (size_t)(mb + row) * SEQ;
  const u16* arow1 = pr + (size_t)(mb + 16 + row) * SEQ;

  for (int kk = 0; kk < SEQ; kk += 32) {
    FragBf a0 = load_frag(arow0 + kk, hi);
    FragBf a1 = load_frag(arow1 + kk, hi);
#pragma unroll
    for (int t = 0; t < 4; ++t) {
      FragBf bfr = load_frag(vb + (size_t)(t * 16 + row) * SEQ + kk, hi);
      acc[0][t] = wmma_bf16(a0, bfr, acc[0][t]);
      acc[1][t] = wmma_bf16(a1, bfr, acc[1][t]);
    }
  }

#pragma unroll
  for (int u = 0; u < 2; ++u)
#pragma unroll
    for (int t = 0; t < 4; ++t)
#pragma unroll
      for (int r = 0; r < 8; ++r) {
        int m = mb + u * 16 + hi * 8 + r;
        int n = t * 16 + row;
        obf[((size_t)(b * SEQ + m)) * DIM + hd * DH + n] = f2bf(acc[u][t][r]);
      }
}

// ---------------------------------------------------------------------------
// SwiGLU: hmid = silu(g[:, :F]) * g[:, F:], bf16 out.
// ---------------------------------------------------------------------------
__global__ __launch_bounds__(256) void k_silu(const float* __restrict__ g,
                                              u16* __restrict__ hmid) {
  size_t idx = (size_t)blockIdx.x * 256 + threadIdx.x;   // M*F elements
  size_t row = idx >> 12;          // / 4096
  size_t j = idx & 4095;
  float a = g[row * (2 * FF) + j];
  float b = g[row * (2 * FF) + FF + j];
  float s = a / (1.0f + __expf(-a));
  hmid[idx] = f2bf(s * b);
}

// ---------------------------------------------------------------------------
// Pooling head: per-row score, per-batch softmax over S, weighted sum, cls.
// ---------------------------------------------------------------------------
__global__ __launch_bounds__(256) void k_score_rows(const float* __restrict__ hf,
                                                    const float* __restrict__ ws,
                                                    float* __restrict__ srow) {
  int rid = blockIdx.x * 256 + threadIdx.x;   // M rows
  const float* hr = hf + (size_t)rid * DIM;
  float s = 0.f;
  for (int d = 0; d < DIM; ++d) s += hr[d] * ws[d];
  srow[rid] = s;
}

__global__ __launch_bounds__(256) void k_softmax_pw(const float* __restrict__ srow,
                                                    float* __restrict__ pw) {
  int b = blockIdx.x;
  const float* sr = srow + (size_t)b * SEQ;
  __shared__ float red[256];
  float vals[4];
  float m = -1e30f;
#pragma unroll
  for (int j = 0; j < 4; ++j) {
    vals[j] = sr[threadIdx.x + j * 256];
    m = fmaxf(m, vals[j]);
  }
  red[threadIdx.x] = m; __syncthreads();
  for (int o = 128; o > 0; o >>= 1) {
    if (threadIdx.x < o) red[threadIdx.x] = fmaxf(red[threadIdx.x], red[threadIdx.x + o]);
    __syncthreads();
  }
  float rmax = red[0]; __syncthreads();
  float sum = 0.f;
#pragma unroll
  for (int j = 0; j < 4; ++j) { vals[j] = __expf(vals[j] - rmax); sum += vals[j]; }
  red[threadIdx.x] = sum; __syncthreads();
  for (int o = 128; o > 0; o >>= 1) {
    if (threadIdx.x < o) red[threadIdx.x] += red[threadIdx.x + o];
    __syncthreads();
  }
  float inv = 1.0f / red[0];
#pragma unroll
  for (int j = 0; j < 4; ++j)
    pw[(size_t)b * SEQ + threadIdx.x + j * 256] = vals[j] * inv;
}

__global__ __launch_bounds__(256) void k_pool(const float* __restrict__ pw,
                                              const float* __restrict__ hf,
                                              float* __restrict__ pooled) {
  int d = blockIdx.x * 256 + threadIdx.x;
  int b = blockIdx.y;
  float s = 0.f;
  for (int t = 0; t < SEQ; ++t)
    s += pw[(size_t)b * SEQ + t] * hf[((size_t)(b * SEQ + t)) * DIM + d];
  pooled[(size_t)b * DIM + d] = s;
}

__global__ __launch_bounds__(256) void k_cls(const float* __restrict__ pooled,
                                             const float* __restrict__ wcls,
                                             float* __restrict__ out) {
  int idx = blockIdx.x * 256 + threadIdx.x;   // B*S = 4096
  int b = idx >> 10;
  int n = idx & 1023;
  float s = 0.f;
  for (int d = 0; d < DIM; ++d)
    s += pooled[(size_t)b * DIM + d] * wcls[(size_t)d * SEQ + n];
  out[idx] = s;
}

// ---------------------------------------------------------------------------
// Host orchestration
// ---------------------------------------------------------------------------
extern "C" void kernel_launch(void* const* d_in, const int* in_sizes, int n_in,
                              void* d_out, int out_size, void* d_ws, size_t ws_size,
                              hipStream_t stream) {
  (void)in_sizes; (void)n_in; (void)out_size; (void)ws_size;

  const int*   tokens  = (const int*)  d_in[0];
  const float* tok_emb = (const float*)d_in[1];
  const float* ln1_s   = (const float*)d_in[2];
  const float* ln1_b   = (const float*)d_in[3];
  const float* w_qkv   = (const float*)d_in[4];
  const float* w_proj  = (const float*)d_in[5];
  const float* ln2_s   = (const float*)d_in[6];
  const float* ln2_b   = (const float*)d_in[7];
  const float* w_in    = (const float*)d_in[8];
  const float* w_out   = (const float*)d_in[9];
  const float* lnf_s   = (const float*)d_in[10];
  const float* lnf_b   = (const float*)d_in[11];
  const float* w_score = (const float*)d_in[12];
  const float* w_cls   = (const float*)d_in[13];

  const int M = MROWS;

  char* p = (char*)d_ws;
  auto carve = [&](size_t bytes) -> void* {
    void* r = (void*)p;
    p += (bytes + 255) & ~(size_t)255;
    return r;
  };

  float* h      = (float*)carve((size_t)M * DIM * 4);                 // 16 MB
  u16*   xbf    = (u16*)  carve((size_t)M * DIM * 2);                 //  8 MB
  float* qkv    = (float*)carve((size_t)M * 3 * DIM * 4);             // 48 MB
  u16*   wbuf   = (u16*)  carve((size_t)DIM * 2 * FF * 2);            // 16 MB
  u16*   qbf    = (u16*)  carve((size_t)BATCH * HEADS * SEQ * DH * 2);//  8 MB
  u16*   kbf    = (u16*)  carve((size_t)BATCH * HEADS * SEQ * DH * 2);//  8 MB
  u16*   vt     = (u16*)  carve((size_t)BATCH * HEADS * DH * SEQ * 2);//  8 MB
  float* scores = (float*)carve((size_t)BATCH * HEADS * SEQ * SEQ * 4);//256 MB
  u16*   probs  = (u16*)  carve((size_t)BATCH * HEADS * SEQ * SEQ * 2);//128 MB
  u16*   obf    = (u16*)  carve((size_t)M * DIM * 2);                 //  8 MB
  float* hf     = (float*)carve((size_t)M * DIM * 4);                 // 16 MB
  float* srow   = (float*)carve((size_t)M * 4);
  float* pw     = (float*)carve((size_t)M * 4);
  float* pooled = (float*)carve((size_t)BATCH * DIM * 4);
  // Phase-disjoint aliases (attention scratch reused by the MLP):
  float* g    = scores;      // (M, 2F) f32 = 128 MB  <= 256 MB
  u16*   hmid = probs;       // (M, F) bf16 =  32 MB  <= 128 MB

  k_embed<<<M, 256, 0, stream>>>(tokens, tok_emb, h);

  for (int l = 0; l < LAYERS; ++l) {
    const float* l1s = ln1_s + (size_t)l * DIM;
    const float* l1b = ln1_b + (size_t)l * DIM;
    const float* l2s = ln2_s + (size_t)l * DIM;
    const float* l2b = ln2_b + (size_t)l * DIM;
    const float* Wqkv = w_qkv + (size_t)l * DIM * 3 * DIM;
    const float* Wp   = w_proj + (size_t)l * DIM * DIM;
    const float* Wi   = w_in  + (size_t)l * DIM * 2 * FF;
    const float* Wo   = w_out + (size_t)l * FF * DIM;

    // ---- attention ----
    k_layernorm<<<M, 256, 0, stream>>>(h, l1s, l1b, xbf, nullptr);

    int kn = DIM * 3 * DIM;
    k_convert<<<(kn + 255) / 256, 256, 0, stream>>>(Wqkv, wbuf, DIM, 3 * DIM);
    k_gemm<<<dim3(3 * DIM / 64, M / 32), 32, 0, stream>>>(xbf, wbuf, nullptr, qkv,
                                                          M, 3 * DIM, DIM);

    k_rope_split<<<(BATCH * SEQ * HEADS * HALF) / 256, 256, 0, stream>>>(qkv, qbf, kbf, vt);

    k_scores<<<dim3(SEQ / 64, SEQ / 16, BATCH * HEADS), 32, 0, stream>>>(qbf, kbf, scores);
    k_softmax_probs<<<BATCH * HEADS * SEQ, 256, 0, stream>>>(scores, probs);
    k_attn_out<<<dim3(SEQ / 32, BATCH * HEADS), 32, 0, stream>>>(probs, vt, obf);

    kn = DIM * DIM;
    k_convert<<<(kn + 255) / 256, 256, 0, stream>>>(Wp, wbuf, DIM, DIM);
    k_gemm<<<dim3(DIM / 64, M / 32), 32, 0, stream>>>(obf, wbuf, h, h, M, DIM, DIM);

    // ---- MLP ----
    k_layernorm<<<M, 256, 0, stream>>>(h, l2s, l2b, xbf, nullptr);

    kn = DIM * 2 * FF;
    k_convert<<<(kn + 255) / 256, 256, 0, stream>>>(Wi, wbuf, DIM, 2 * FF);
    k_gemm<<<dim3(2 * FF / 64, M / 32), 32, 0, stream>>>(xbf, wbuf, nullptr, g,
                                                         M, 2 * FF, DIM);

    k_silu<<<(int)(((size_t)M * FF) / 256), 256, 0, stream>>>(g, hmid);

    kn = FF * DIM;
    k_convert<<<(kn + 255) / 256, 256, 0, stream>>>(Wo, wbuf, FF, DIM);
    k_gemm<<<dim3(DIM / 64, M / 32), 32, 0, stream>>>(hmid, wbuf, h, h, M, DIM, FF);
  }

  // ---- pooling head ----
  k_layernorm<<<M, 256, 0, stream>>>(h, lnf_s, lnf_b, nullptr, hf);
  k_score_rows<<<M / 256, 256, 0, stream>>>(hf, w_score, srow);
  k_softmax_pw<<<BATCH, 256, 0, stream>>>(srow, pw);
  k_pool<<<dim3(DIM / 256, BATCH), 256, 0, stream>>>(pw, hf, pooled);
  k_cls<<<(BATCH * SEQ) / 256, 256, 0, stream>>>(pooled, w_cls, (float*)d_out);
}
